// VisionAttention_14671608283774
// MI455X (gfx1250) — matrix operand order repaired and checked by reading, not compiled, not run on hardware
//
#include <hip/hip_runtime.h>
#include <hip/hip_bf16.h>

#define N_TOK 4096
#define NHEAD 12
#define HD    64
#define DMODEL 768
#define SCALE 0.125f

typedef __attribute__((ext_vector_type(16))) _Float16 v16h;
typedef __attribute__((ext_vector_type(8)))  _Float16 v8h;
typedef __attribute__((ext_vector_type(8)))  float    v8f;

// ---------------------------------------------------------------------------
// WMMA fragment helpers (layouts per CDNA5 ISA 7.12.2, wave32)
// ---------------------------------------------------------------------------

// A-matrix 16x32 f16: lane m=l&15 holds row m.
//   halves 0..7  -> K = c0 + (l>>4)*8 + i
//   halves 8..15 -> K = c0 + 16 + (l>>4)*8 + i
__device__ __forceinline__ v16h load_a_frag(const _Float16* base, int ldm, int lane) {
    const int m  = lane & 15;
    const int hi = (lane >> 4) & 1;
    const _Float16* p = base + m * ldm + hi * 8;
    v8h lo = *(const v8h*)(p);
    v8h hh = *(const v8h*)(p + 16);
    v16h a;
#pragma unroll
    for (int i = 0; i < 8; ++i) { a[i] = lo[i]; a[i + 8] = hh[i]; }
    return a;
}

// B-matrix 32x16 f16: lane n=l&15 holds column n.
//   halves 0..7  -> K = c0 + (l>>4)*8 + i ; halves 8..15 -> K = c0 + 16 + (l>>4)*8 + i
// src is laid out [n][k] row-major with leading dim ldm.
__device__ __forceinline__ v16h load_b_frag(const _Float16* base, int ldm, int lane) {
    const int n  = lane & 15;
    const int hi = (lane >> 4) & 1;
    const _Float16* p = base + n * ldm + hi * 8;
    v8h lo = *(const v8h*)(p);
    v8h hh = *(const v8h*)(p + 16);
    v16h b;
#pragma unroll
    for (int i = 0; i < 8; ++i) { b[i] = lo[i]; b[i + 8] = hh[i]; }
    return b;
}

__device__ __forceinline__ v8f wmma_f16(v16h a, v16h b, v8f c) {
    return __builtin_amdgcn_wmma_f32_16x16x32_f16(
        /*neg_a=*/false, a, /*neg_b=*/false, b,
        /*c_mod=*/(short)0, c, /*reuse_a=*/false, /*reuse_b=*/false);
}

// ---------------------------------------------------------------------------
// Kernel 1: fp32 -> f16 conversion
// ---------------------------------------------------------------------------
__global__ void va_cvt_f16(const float* __restrict__ src, _Float16* __restrict__ dst, int count) {
    int i = (blockIdx.x * 256 + threadIdx.x) * 4;
    if (i < count) {
#pragma unroll
        for (int j = 0; j < 4; ++j) dst[i + j] = (_Float16)src[i + j];
    }
}

// ---------------------------------------------------------------------------
// Kernel 2: generic f16 GEMM, fp32 accumulate, + bias.
// C[M][Nn] = A[M][K] * B[K][Nn] + bias.  Block tile 64x64, 128 threads (4 waves),
// wave (wm,wn) computes a 32x32 subtile = 2x2 WMMA C tiles.  K step = 32.
// ---------------------------------------------------------------------------
__global__ __launch_bounds__(128) void va_gemm_f16(
    const _Float16* __restrict__ A, const _Float16* __restrict__ B,
    const float* __restrict__ bias, float* __restrict__ C,
    int M, int K, int Nn) {
    __shared__ __align__(16) _Float16 ldsA[64][40];   // [row][k], pad 8 -> 80B rows
    __shared__ __align__(16) _Float16 ldsB[64][40];   // [col][k] (transposed)

    const int t = threadIdx.x;
    const int w = t >> 5, lane = t & 31;
    const int m = lane & 15, hi = (lane >> 4) & 1;
    const int wm = w >> 1, wn = w & 1;
    const int rowBase = blockIdx.x * 64;
    const int colBase = blockIdx.y * 64;

    v8f acc[2][2];
#pragma unroll
    for (int i = 0; i < 2; ++i)
#pragma unroll
        for (int j = 0; j < 2; ++j)
#pragma unroll
            for (int r = 0; r < 8; ++r) acc[i][j][r] = 0.0f;

    for (int kk = 0; kk < K; kk += 32) {
        // stage A tile 64x32
#pragma unroll
        for (int i = 0; i < 2; ++i) {
            int lin = (i * 128 + t) * 8;
            int r = lin >> 5, k0 = lin & 31;
            *(v8h*)(&ldsA[r][k0]) =
                *(const v8h*)(A + (size_t)(rowBase + r) * K + kk + k0);
        }
        // stage B tile 32x64, transposed into [n][k]
#pragma unroll
        for (int i = 0; i < 2; ++i) {
            int lin = (i * 128 + t) * 8;
            int kr = lin >> 6, n0 = lin & 63;
            v8h bv = *(const v8h*)(B + (size_t)(kk + kr) * Nn + colBase + n0);
#pragma unroll
            for (int j = 0; j < 8; ++j) ldsB[n0 + j][kr] = bv[j];
        }
        __syncthreads();

        v16h af0 = load_a_frag(&ldsA[wm * 32][0],      40, lane);
        v16h af1 = load_a_frag(&ldsA[wm * 32 + 16][0], 40, lane);
        v16h bf0 = load_b_frag(&ldsB[wn * 32][0],      40, lane);
        v16h bf1 = load_b_frag(&ldsB[wn * 32 + 16][0], 40, lane);
        acc[0][0] = wmma_f16(af0, bf0, acc[0][0]);
        acc[0][1] = wmma_f16(af0, bf1, acc[0][1]);
        acc[1][0] = wmma_f16(af1, bf0, acc[1][0]);
        acc[1][1] = wmma_f16(af1, bf1, acc[1][1]);
        __syncthreads();
    }

#pragma unroll
    for (int i = 0; i < 2; ++i)
#pragma unroll
        for (int j = 0; j < 2; ++j)
#pragma unroll
            for (int r = 0; r < 8; ++r) {
                int row = rowBase + wm * 32 + i * 16 + r + hi * 8;
                int col = colBase + wn * 32 + j * 16 + m;
                C[(size_t)row * Nn + col] = acc[i][j][r] + bias[col];
            }
}

// ---------------------------------------------------------------------------
// Kernel 3: RoPE epilogue.  qkv fp32 [n][3][H][64] -> head-major f16 Q/K/V.
// q' = q*c + rotate_half(q)*s ; q scaled by SCALE.
// ---------------------------------------------------------------------------
__global__ void va_rope(const float* __restrict__ qkv,
                        const float* __restrict__ cs, const float* __restrict__ sn,
                        _Float16* __restrict__ Qh, _Float16* __restrict__ Kh,
                        _Float16* __restrict__ Vh) {
    int idx = blockIdx.x * 256 + threadIdx.x;          // N*H*32 pairs
    if (idx >= N_TOK * NHEAD * 32) return;
    int d = idx & 31;
    int h = (idx >> 5) % NHEAD;
    int n = idx / (32 * NHEAD);

    const float* base = qkv + (size_t)n * (3 * DMODEL);
    float c0 = cs[n * HD + d],      c1 = cs[n * HD + d + 32];
    float s0 = sn[n * HD + d],      s1 = sn[n * HD + d + 32];

    float q0 = base[h * HD + d],                q1 = base[h * HD + d + 32];
    float k0 = base[DMODEL + h * HD + d],       k1 = base[DMODEL + h * HD + d + 32];
    float v0 = base[2 * DMODEL + h * HD + d],   v1 = base[2 * DMODEL + h * HD + d + 32];

    float qo0 = (q0 * c0 - q1 * s0) * SCALE;
    float qo1 = (q1 * c1 + q0 * s1) * SCALE;
    float ko0 = k0 * c0 - k1 * s0;
    float ko1 = k1 * c1 + k0 * s1;

    size_t ob = (size_t)h * N_TOK * HD + (size_t)n * HD + d;
    Qh[ob] = (_Float16)qo0;  Qh[ob + 32] = (_Float16)qo1;
    Kh[ob] = (_Float16)ko0;  Kh[ob + 32] = (_Float16)ko1;
    Vh[ob] = (_Float16)v0;   Vh[ob + 32] = (_Float16)v1;
}

// ---------------------------------------------------------------------------
// Kernel 4: flash attention.  Grid (N/64, H).  128 threads = 4 waves; each
// wave owns 16 query rows.  KV processed in 64-key blocks via LDS; online
// softmax; P staged through per-wave LDS to feed A fragments of P@V.
// ---------------------------------------------------------------------------
__global__ __launch_bounds__(128) void va_attn(
    const _Float16* __restrict__ Qh, const _Float16* __restrict__ Kh,
    const _Float16* __restrict__ Vh, _Float16* __restrict__ Oh) {
    __shared__ __align__(16) _Float16 ldsK[64][72];       // [key][d]
    __shared__ __align__(16) _Float16 ldsVt[64][72];      // [d][key]
    __shared__ __align__(16) _Float16 ldsP[4][16][72];    // per-wave P tile [row][key]

    const int h = blockIdx.y;
    const int qblk = blockIdx.x;
    const int t = threadIdx.x;
    const int w = t >> 5, lane = t & 31;
    const int m = lane & 15, hi = (lane >> 4) & 1;

    const _Float16* Qbase = Qh + ((size_t)h * N_TOK + qblk * 64 + w * 16) * HD;
    const _Float16* Kbase = Kh + (size_t)h * N_TOK * HD;
    const _Float16* Vbase = Vh + (size_t)h * N_TOK * HD;

    // Q rows of this wave held as 2 A fragments (K-dim 0..31 and 32..63)
    v16h qa0 = load_a_frag(Qbase, HD, lane);
    v16h qa1 = load_a_frag(Qbase + 32, HD, lane);

    float mi[8], li[8];
    v8f o[4];
#pragma unroll
    for (int r = 0; r < 8; ++r) { mi[r] = -1e30f; li[r] = 0.0f; }
#pragma unroll
    for (int nt = 0; nt < 4; ++nt)
#pragma unroll
        for (int r = 0; r < 8; ++r) o[nt][r] = 0.0f;

    for (int kb = 0; kb < N_TOK / 64; ++kb) {
        // cooperative stage of K (row-major) and V (transposed) blocks
#pragma unroll
        for (int i = 0; i < 4; ++i) {
            int lin = (i * 128 + t) * 8;
            int key = lin >> 6, d0 = lin & 63;
            v8h kv = *(const v8h*)(Kbase + (size_t)(kb * 64 + key) * HD + d0);
            *(v8h*)(&ldsK[key][d0]) = kv;
            v8h vv = *(const v8h*)(Vbase + (size_t)(kb * 64 + key) * HD + d0);
#pragma unroll
            for (int j = 0; j < 8; ++j) ldsVt[d0 + j][key] = vv[j];
        }
        __syncthreads();

        // S = Q * K^T : 4 column tiles x (K-dim 64 = 2 WMMA)
        v8f s[4];
#pragma unroll
        for (int ct = 0; ct < 4; ++ct) {
#pragma unroll
            for (int r = 0; r < 8; ++r) s[ct][r] = 0.0f;
            v16h b0 = load_b_frag(&ldsK[ct * 16][0],  72, lane);
            s[ct] = wmma_f16(qa0, b0, s[ct]);
            v16h b1 = load_b_frag(&ldsK[ct * 16][32], 72, lane);
            s[ct] = wmma_f16(qa1, b1, s[ct]);
        }

        // online softmax (row = r + hi*8, col = ct*16 + m); row reductions stay
        // inside the 16-lane half that owns the row (shfl_xor masks 1,2,4,8).
#pragma unroll
        for (int r = 0; r < 8; ++r) {
            float mnew = mi[r];
#pragma unroll
            for (int ct = 0; ct < 4; ++ct) mnew = fmaxf(mnew, s[ct][r]);
#pragma unroll
            for (int msk = 1; msk < 16; msk <<= 1)
                mnew = fmaxf(mnew, __shfl_xor(mnew, msk));
            float corr = __expf(mi[r] - mnew);
            mi[r] = mnew;
            float rs = 0.0f;
#pragma unroll
            for (int ct = 0; ct < 4; ++ct) {
                float p = __expf(s[ct][r] - mnew);
                s[ct][r] = p;
                rs += p;
            }
#pragma unroll
            for (int msk = 1; msk < 16; msk <<= 1) rs += __shfl_xor(rs, msk);
            li[r] = li[r] * corr + rs;
#pragma unroll
            for (int nt = 0; nt < 4; ++nt) o[nt][r] *= corr;
        }

        // stage P (f16) into per-wave LDS in natural [row][key] order
#pragma unroll
        for (int ct = 0; ct < 4; ++ct)
#pragma unroll
            for (int r = 0; r < 8; ++r)
                ldsP[w][r + hi * 8][ct * 16 + m] = (_Float16)s[ct][r];

        // per-wave LDS is in-order on CDNA5; fence so reads see the P stores
        asm volatile("s_wait_dscnt 0" ::: "memory");

        // O += P * V : A frags from ldsP, B frags from transposed V
        v16h pa0 = load_a_frag(&ldsP[w][0][0],  72, lane);
        v16h pa1 = load_a_frag(&ldsP[w][0][32], 72, lane);
#pragma unroll
        for (int nt = 0; nt < 4; ++nt) {
            v16h b0 = load_b_frag(&ldsVt[nt * 16][0],  72, lane);
            o[nt] = wmma_f16(pa0, b0, o[nt]);
            v16h b1 = load_b_frag(&ldsVt[nt * 16][32], 72, lane);
            o[nt] = wmma_f16(pa1, b1, o[nt]);
        }
        __syncthreads();
    }

    // normalize and write back as f16 [n][h*64+d] for the projection GEMM
#pragma unroll
    for (int r = 0; r < 8; ++r) {
        float inv = 1.0f / li[r];
        int row = qblk * 64 + w * 16 + r + hi * 8;
#pragma unroll
        for (int nt = 0; nt < 4; ++nt) {
            int col = h * HD + nt * 16 + m;
            Oh[(size_t)row * DMODEL + col] = (_Float16)(o[nt][r] * inv);
        }
    }
}

// ---------------------------------------------------------------------------
// Host launcher
// ---------------------------------------------------------------------------
extern "C" void kernel_launch(void* const* d_in, const int* in_sizes, int n_in,
                              void* d_out, int out_size, void* d_ws, size_t ws_size,
                              hipStream_t stream) {
    const float* hidden = (const float*)d_in[0];   // [4096][768]
    const float* cs     = (const float*)d_in[1];   // [4096][64]
    const float* sn     = (const float*)d_in[2];   // [4096][64]
    const float* qkv_w  = (const float*)d_in[3];   // [768][2304]
    const float* qkv_b  = (const float*)d_in[4];   // [2304]
    const float* proj_w = (const float*)d_in[5];   // [768][768]
    const float* proj_b = (const float*)d_in[6];   // [768]
    float* out = (float*)d_out;

    char* ws = (char*)d_ws;
    size_t off = 0;
    _Float16* Xh     = (_Float16*)(ws + off); off += (size_t)N_TOK * DMODEL * 2;       // 6.3 MB
    _Float16* Wqkvh  = (_Float16*)(ws + off); off += (size_t)DMODEL * 3 * DMODEL * 2;  // 3.5 MB
    _Float16* Wprojh = (_Float16*)(ws + off); off += (size_t)DMODEL * DMODEL * 2;      // 1.2 MB
    float*    qkvF   = (float*)   (ws + off); off += (size_t)N_TOK * 3 * DMODEL * 4;   // 37.7 MB
    _Float16* Qh     = (_Float16*)(ws + off); off += (size_t)N_TOK * DMODEL * 2;
    _Float16* Kh     = (_Float16*)(ws + off); off += (size_t)N_TOK * DMODEL * 2;
    _Float16* Vh     = (_Float16*)(ws + off); off += (size_t)N_TOK * DMODEL * 2;
    _Float16* Ah     = (_Float16*)(ws + off); off += (size_t)N_TOK * DMODEL * 2;

    // 1) convert to f16
    va_cvt_f16<<<(N_TOK * DMODEL) / 1024, 256, 0, stream>>>(hidden, Xh, N_TOK * DMODEL);
    va_cvt_f16<<<(DMODEL * 3 * DMODEL) / 1024, 256, 0, stream>>>(qkv_w, Wqkvh, DMODEL * 3 * DMODEL);
    va_cvt_f16<<<(DMODEL * DMODEL) / 1024, 256, 0, stream>>>(proj_w, Wprojh, DMODEL * DMODEL);

    // 2) QKV GEMM: [4096x768] x [768x2304] + bias
    va_gemm_f16<<<dim3(N_TOK / 64, (3 * DMODEL) / 64), 128, 0, stream>>>(
        Xh, Wqkvh, qkv_b, qkvF, N_TOK, DMODEL, 3 * DMODEL);

    // 3) RoPE + head-major repack
    va_rope<<<(N_TOK * NHEAD * 32) / 256, 256, 0, stream>>>(qkvF, cs, sn, Qh, Kh, Vh);

    // 4) flash attention per (64-row qblock, head)
    va_attn<<<dim3(N_TOK / 64, NHEAD), 128, 0, stream>>>(Qh, Kh, Vh, Ah);

    // 5) projection GEMM: [4096x768] x [768x768] + bias -> fp32 out
    va_gemm_f16<<<dim3(N_TOK / 64, DMODEL / 64), 128, 0, stream>>>(
        Ah, Wprojh, proj_b, out, N_TOK, DMODEL, DMODEL);
}
